// DemixedDilatedTransformerModel_6493990552269
// MI455X (gfx1250) — compile-verified, matrix operand order checked by reading, not run on hardware
//
#include <hip/hip_runtime.h>
#include <math.h>

// ---------------------------------------------------------------------------
// DemixedDilatedTransformerModel for MI455X (gfx1250), wave32 + WMMA f16.
// All GEMMs (QKV, attn out-proj, FFNs, instrument encoder) run through one
// LDS-staged WMMA kernel: 256-thread block -> 128x64 C macro-tile, 8 waves,
// each wave 2x2 v_wmma_f32_16x16x32_f16 tiles, K fully unrolled (K=128/512).
// ---------------------------------------------------------------------------

typedef __attribute__((ext_vector_type(16))) _Float16 v16h;
typedef __attribute__((ext_vector_type(8)))  _Float16 v8h;
typedef __attribute__((ext_vector_type(8)))  float    v8f;

#define BATCH 2
#define INSTR 5
#define BI    10          // BATCH*INSTR
#define TLEN  2048
#define DMODEL 128
#define NHEAD 8
#define HDIM  16
#define NATTN 5
#define MTOK  (BI*TLEN)   // 20480 rows for all big GEMMs
#define SCALE_INV 0.25f   // 1/sqrt(HD)=1/4

// ---------------- f32 -> f16 packing helpers --------------------------------
__device__ __forceinline__ v16h pack16(float4 a0, float4 a1, float4 a2, float4 a3) {
  v16h r;
  r[0]=(_Float16)a0.x;  r[1]=(_Float16)a0.y;  r[2]=(_Float16)a0.z;  r[3]=(_Float16)a0.w;
  r[4]=(_Float16)a1.x;  r[5]=(_Float16)a1.y;  r[6]=(_Float16)a1.z;  r[7]=(_Float16)a1.w;
  r[8]=(_Float16)a2.x;  r[9]=(_Float16)a2.y;  r[10]=(_Float16)a2.z; r[11]=(_Float16)a2.w;
  r[12]=(_Float16)a3.x; r[13]=(_Float16)a3.y; r[14]=(_Float16)a3.z; r[15]=(_Float16)a3.w;
  return r;
}
__device__ __forceinline__ v8h pack8(float4 a0, float4 a1) {
  v8h r;
  r[0]=(_Float16)a0.x; r[1]=(_Float16)a0.y; r[2]=(_Float16)a0.z; r[3]=(_Float16)a0.w;
  r[4]=(_Float16)a1.x; r[5]=(_Float16)a1.y; r[6]=(_Float16)a1.z; r[7]=(_Float16)a1.w;
  return r;
}

// Fragment gather from LDS in the ISA 16-bit A/B layout:
// lanes 0-15 get K 0-7 / 16-23, lanes 16-31 get K 8-15 / 24-31 (hl selects).
#define BKP 48   // padded LDS row stride (halves): breaks 4-way bank conflicts
__device__ __forceinline__ v16h load_frag(const _Float16* S, int row, int hl) {
  const v8h lo = *reinterpret_cast<const v8h*>(&S[row * BKP + hl * 8]);
  const v8h hi = *reinterpret_cast<const v8h*>(&S[row * BKP + 16 + hl * 8]);
  v16h r;
  #pragma unroll
  for (int j = 0; j < 8; ++j) { r[j] = lo[j]; r[j + 8] = hi[j]; }
  return r;
}

// Epilogue for one 16x16 tile: bias + {none,relu,gelu} + optional residual.
__device__ __forceinline__ void store_tile(float* C, const float* residual,
                                           const float* __restrict__ bias, v8f acc,
                                           int gm0, int gn0, int N, int act,
                                           int hl, int l16) {
  const int gn = gn0 + l16;
  const float bc = bias ? bias[gn] : 0.0f;
  #pragma unroll
  for (int r = 0; r < 8; ++r) {          // C layout: row = r + 8*hl, col = l16
    const int gm = gm0 + (hl << 3) + r;
    float v = acc[r] + bc;
    if (act == 1)      v = fmaxf(v, 0.0f);
    else if (act == 2) v = 0.5f * v * (1.0f + erff(v * 0.7071067811865475f));
    if (residual)      v += residual[(size_t)gm * N + gn];
    C[(size_t)gm * N + gn] = v;
  }
}

// ---- LDS-staged WMMA GEMM: C[M,N] = post(A[M,K] @ W[N,K]^T + bias) (+res) --
// Block 256 threads = 8 waves; macro-tile 128(M) x 64(N); BK=32.
// Requires M%128==0, N%64==0; K is a compile-time constant (128 or 512).
template <int K>
__global__ void __launch_bounds__(256)
gemm_wmma_kernel(const float* __restrict__ A, const float* __restrict__ W,
                 const float* __restrict__ bias, const float* residual,
                 float* C, int M, int N, int act)
{
  constexpr int BM = 128, BN = 64, BK = 32;
  __shared__ _Float16 As[BM * BKP];     // [row][k] padded
  __shared__ _Float16 Bs[BN * BKP];     // [col(n)][k] padded

  const int tid  = threadIdx.x;
  const int wave = tid >> 5;
  const int lane = tid & 31;
  const int hl   = lane >> 4;
  const int l16  = lane & 15;
  const int wm   = (wave & 3) << 5;     // wave M offset inside macro-tile
  const int wn   = (wave >> 2) << 5;    // wave N offset inside macro-tile
  const int m0   = blockIdx.y * BM;
  const int n0   = blockIdx.x * BN;

  // staging assignment: A -> 2 threads/row x 16 floats; B -> 4 threads/row x 8
  const int ar = tid >> 1, ac = (tid & 1) << 4;
  const int br = tid >> 2, bc = (tid & 3) << 3;
  const float* Aload = A + (size_t)(m0 + ar) * K + ac;
  const float* Bload = W + (size_t)(n0 + br) * K + bc;

  v8f acc00 = {0.f,0.f,0.f,0.f,0.f,0.f,0.f,0.f};
  v8f acc01 = acc00, acc10 = acc00, acc11 = acc00;

  #pragma unroll
  for (int k0 = 0; k0 < K; k0 += BK) {
    // stage A (128x32) and B (64x32) as f16 into LDS
    float4 a0 = *reinterpret_cast<const float4*>(Aload + k0);
    float4 a1 = *reinterpret_cast<const float4*>(Aload + k0 + 4);
    float4 a2 = *reinterpret_cast<const float4*>(Aload + k0 + 8);
    float4 a3 = *reinterpret_cast<const float4*>(Aload + k0 + 12);
    *reinterpret_cast<v16h*>(&As[ar * BKP + ac]) = pack16(a0, a1, a2, a3);
    float4 b0 = *reinterpret_cast<const float4*>(Bload + k0);
    float4 b1 = *reinterpret_cast<const float4*>(Bload + k0 + 4);
    *reinterpret_cast<v8h*>(&Bs[br * BKP + bc]) = pack8(b0, b1);
    if (k0 + BK < K) {                     // gfx1250 global_prefetch_b8 path
      __builtin_prefetch(Aload + k0 + BK, 0, 1);
      __builtin_prefetch(Bload + k0 + BK, 0, 1);
    }
    __syncthreads();

    // fragments from LDS, 2x2 WMMA per wave (A/B fragments each reused twice)
    const v16h aF0 = load_frag(As, wm + l16,      hl);
    const v16h aF1 = load_frag(As, wm + 16 + l16, hl);
    const v16h bF0 = load_frag(Bs, wn + l16,      hl);
    const v16h bF1 = load_frag(Bs, wn + 16 + l16, hl);
    acc00 = __builtin_amdgcn_wmma_f32_16x16x32_f16(false, aF0, false, bF0, (short)0, acc00, false, false);
    acc01 = __builtin_amdgcn_wmma_f32_16x16x32_f16(false, aF0, false, bF1, (short)0, acc01, false, false);
    acc10 = __builtin_amdgcn_wmma_f32_16x16x32_f16(false, aF1, false, bF0, (short)0, acc10, false, false);
    acc11 = __builtin_amdgcn_wmma_f32_16x16x32_f16(false, aF1, false, bF1, (short)0, acc11, false, false);
    __syncthreads();
  }

  store_tile(C, residual, bias, acc00, m0 + wm,      n0 + wn,      N, act, hl, l16);
  store_tile(C, residual, bias, acc01, m0 + wm,      n0 + wn + 16, N, act, hl, l16);
  store_tile(C, residual, bias, acc10, m0 + wm + 16, n0 + wn,      N, act, hl, l16);
  store_tile(C, residual, bias, acc11, m0 + wm + 16, n0 + wn + 16, N, act, hl, l16);
}

// ---------------- LayerNorm over D=128, one block per row, LDS reduction ----
__global__ void __launch_bounds__(128)
layernorm_kernel(const float* __restrict__ x, const float* __restrict__ s,
                 const float* __restrict__ b, float* __restrict__ y)
{
  __shared__ float red[128];
  const size_t row = blockIdx.x;
  const int i = threadIdx.x;
  const float v = x[row * DMODEL + i];
  red[i] = v;
  __syncthreads();
  for (int o = 64; o > 0; o >>= 1) { if (i < o) red[i] += red[i + o]; __syncthreads(); }
  const float m = red[0] * (1.0f / DMODEL);
  __syncthreads();
  const float dv = v - m;
  red[i] = dv * dv;
  __syncthreads();
  for (int o = 64; o > 0; o >>= 1) { if (i < o) red[i] += red[i + o]; __syncthreads(); }
  const float var = red[0] * (1.0f / DMODEL);
  y[row * DMODEL + i] = dv * rsqrtf(var + 1e-5f) * s[i] + b[i];
}

// ---------------- Dilated attention core (per b,h,t thread) -----------------
__global__ void __launch_bounds__(256)
dilated_attn_kernel(const float* __restrict__ q, const float* __restrict__ k,
                    const float* __restrict__ v, const float* __restrict__ Erl,
                    float* __restrict__ sa, int layer)
{
  const int idx = blockIdx.x * 256 + threadIdx.x;
  if (idx >= BI * NHEAD * TLEN) return;
  const int t  = idx % TLEN;
  const int h  = (idx / TLEN) % NHEAD;
  const int bi = idx / (TLEN * NHEAD);
  const int dil = 1 << layer;
  const int P   = 2 * dil;
  const int L   = TLEN + 2 * P;
  const int kh  = (h == 7) ? 6 : h;   // reference reuses head 6's k for head 7
  const int vh  = h;
  int shift;
  if (h < 4) shift = 0; else if (h == 4) shift = -2; else if (h == 5) shift = -1;
  else if (h == 6) shift = 1; else shift = 2;

  float qv[HDIM];
  const float* qp = q + ((size_t)bi * TLEN + t) * DMODEL + h * HDIM;
  #pragma unroll
  for (int j = 0; j < HDIM; ++j) qv[j] = qp[j];
  const float* Eh = Erl + h * HDIM * NATTN;

  float sc[NATTN]; int srcs[NATTN]; float vld[NATTN];
  float mx = -INFINITY;
  #pragma unroll
  for (int a = 0; a < NATTN; ++a) {
    int u = t + dil * (shift + a) + L;   // +L keeps it non-negative
    u %= L;
    int src = u - P;
    src = src < 0 ? 0 : (src > TLEN - 1 ? TLEN - 1 : src);
    const bool valid = (u >= P) && (u < TLEN + P);
    srcs[a] = src;
    vld[a]  = valid ? 1.0f : 0.0f;
    float qk = 0.0f;
    if (valid) {
      const float* kp = k + ((size_t)bi * TLEN + src) * DMODEL + kh * HDIM;
      #pragma unroll
      for (int j = 0; j < HDIM; ++j) qk += qv[j] * kp[j];
    }
    float er = 0.0f;
    #pragma unroll
    for (int j = 0; j < HDIM; ++j) er += qv[j] * Eh[j * NATTN + a];
    float s = (qk + er) * SCALE_INV;
    if (qk == 0.0f) s = -INFINITY;       // mask = where(qk==0, -inf, 0)
    sc[a] = s;
    mx = fmaxf(mx, s);
  }
  float es[NATTN]; float den = 0.0f;
  #pragma unroll
  for (int a = 0; a < NATTN; ++a) { es[a] = expf(sc[a] - mx); den += es[a]; }
  const float rden = 1.0f / den;
  float ov[HDIM];
  #pragma unroll
  for (int j = 0; j < HDIM; ++j) ov[j] = 0.0f;
  #pragma unroll
  for (int a = 0; a < NATTN; ++a) {
    const float w = es[a] * rden * vld[a];
    if (w != 0.0f) {
      const float* vp = v + ((size_t)bi * TLEN + srcs[a]) * DMODEL + vh * HDIM;
      #pragma unroll
      for (int j = 0; j < HDIM; ++j) ov[j] += w * vp[j];
    }
  }
  float* op = sa + ((size_t)bi * TLEN + t) * DMODEL + h * HDIM;
  #pragma unroll
  for (int j = 0; j < HDIM; ++j) op[j] = ov[j];
}

// ---------------- Instrument attention (seq len 5, per n,h,s1 thread) -------
__global__ void __launch_bounds__(256)
instr_attn_kernel(const float* __restrict__ qkv, float* __restrict__ o)
{
  const int idx = blockIdx.x * 256 + threadIdx.x;
  if (idx >= (BATCH * TLEN) * NHEAD * INSTR) return;
  const int s1 = idx % INSTR;
  const int h  = (idx / INSTR) % NHEAD;
  const int n  = idx / (INSTR * NHEAD);
  const float* base = qkv + (size_t)n * INSTR * (3 * DMODEL);
  float qv[HDIM];
  const float* qp = base + s1 * (3 * DMODEL) + h * HDIM;
  #pragma unroll
  for (int j = 0; j < HDIM; ++j) qv[j] = qp[j];
  float sc[INSTR]; float mx = -INFINITY;
  #pragma unroll
  for (int s2 = 0; s2 < INSTR; ++s2) {
    const float* kp = base + s2 * (3 * DMODEL) + DMODEL + h * HDIM;
    float d = 0.0f;
    #pragma unroll
    for (int j = 0; j < HDIM; ++j) d += qv[j] * kp[j];
    sc[s2] = d * SCALE_INV;
    mx = fmaxf(mx, sc[s2]);
  }
  float es[INSTR]; float den = 0.0f;
  #pragma unroll
  for (int s2 = 0; s2 < INSTR; ++s2) { es[s2] = expf(sc[s2] - mx); den += es[s2]; }
  const float rden = 1.0f / den;
  float ov[HDIM];
  #pragma unroll
  for (int j = 0; j < HDIM; ++j) ov[j] = 0.0f;
  #pragma unroll
  for (int s2 = 0; s2 < INSTR; ++s2) {
    const float w = es[s2] * rden;
    const float* vp = base + s2 * (3 * DMODEL) + 2 * DMODEL + h * HDIM;
    #pragma unroll
    for (int j = 0; j < HDIM; ++j) ov[j] += w * vp[j];
  }
  float* op = o + ((size_t)n * INSTR + s1) * DMODEL + h * HDIM;
  #pragma unroll
  for (int j = 0; j < HDIM; ++j) op[j] = ov[j];
}

// ---------------- Conv frontend: conv + bias + maxpool(w=3) + relu ----------
__global__ void __launch_bounds__(256)
conv1_kernel(const float* __restrict__ x, const float* __restrict__ w,
             const float* __restrict__ b, float* __restrict__ out)
{
  const int W1 = 42;
  const int idx = blockIdx.x * 256 + threadIdx.x;
  if (idx >= BI * 32 * TLEN * W1) return;
  const int wp = idx % W1;
  const int t  = (idx / W1) % TLEN;
  const int oc = (idx / (W1 * TLEN)) % 32;
  const int bi = idx / (W1 * TLEN * 32);
  const float* xin = x + (size_t)bi * TLEN * 128;
  const float* wk  = w + oc * 15;            // (32,1,5,3)
  float best = -INFINITY;
  #pragma unroll
  for (int pw = 0; pw < 3; ++pw) {
    const int w0 = wp * 3 + pw;
    float acc = b[oc];
    #pragma unroll
    for (int kh = 0; kh < 5; ++kh) {
      const int tt = t + kh - 2;
      if (tt < 0 || tt >= TLEN) continue;
      const float* xr = xin + (size_t)tt * 128 + w0;
      const float* wr = wk + kh * 3;
      acc += xr[0]*wr[0] + xr[1]*wr[1] + xr[2]*wr[2];
    }
    best = fmaxf(best, acc);
  }
  out[idx] = fmaxf(best, 0.0f);
}

__global__ void __launch_bounds__(256)
conv2_kernel(const float* __restrict__ c1, const float* __restrict__ w,
             const float* __restrict__ b, float* __restrict__ out)
{
  const int W2 = 10;
  const int idx = blockIdx.x * 256 + threadIdx.x;
  if (idx >= BI * 64 * TLEN * W2) return;
  const int wp = idx % W2;
  const int t  = (idx / W2) % TLEN;
  const int oc = (idx / (W2 * TLEN)) % 64;
  const int bi = idx / (W2 * TLEN * 64);
  float best = -INFINITY;
  for (int pw = 0; pw < 3; ++pw) {
    const int w0 = wp * 3 + pw;
    float acc = b[oc];
    for (int ic = 0; ic < 32; ++ic) {
      const float* xr = c1 + (((size_t)(bi * 32 + ic) * TLEN) + t) * 42 + w0;
      const float* wr = w + (oc * 32 + ic) * 12;   // (64,32,1,12)
      #pragma unroll
      for (int kw = 0; kw < 12; ++kw) acc += xr[kw] * wr[kw];
    }
    best = fmaxf(best, acc);
  }
  out[idx] = fmaxf(best, 0.0f);
}

__global__ void __launch_bounds__(256)
conv3_kernel(const float* __restrict__ c2, const float* __restrict__ w,
             const float* __restrict__ b, float* __restrict__ h)
{
  const int idx = blockIdx.x * 256 + threadIdx.x;
  if (idx >= BI * TLEN * DMODEL) return;
  const int oc = idx % DMODEL;
  const int t  = (idx / DMODEL) % TLEN;
  const int bi = idx / (DMODEL * TLEN);
  float best = -INFINITY;
  for (int pw = 0; pw < 3; ++pw) {               // pooled width 1 covers w=0..2
    float acc = b[oc];
    for (int kh = 0; kh < 3; ++kh) {
      const int tt = t + kh - 1;
      if (tt < 0 || tt >= TLEN) continue;
      for (int ic = 0; ic < 64; ++ic) {
        const float* xr = c2 + (((size_t)(bi * 64 + ic) * TLEN) + tt) * 10 + pw;
        const float* wr = w + ((oc * 64 + ic) * 3 + kh) * 6;   // (128,64,3,6)
        #pragma unroll
        for (int kw = 0; kw < 6; ++kw) acc += xr[kw] * wr[kw];
      }
    }
    best = fmaxf(best, acc);
  }
  h[idx] = fmaxf(best, 0.0f);                    // h[bi][t][d]
}

// ---------------- small elementwise / permute / head kernels ----------------
__global__ void zero_kernel(float* p, int n) {
  int i = blockIdx.x * 256 + threadIdx.x;
  if (i < n) p[i] = 0.0f;
}

__global__ void add_inplace_kernel(float* a, const float* __restrict__ b, int n4) {
  int i = blockIdx.x * 256 + threadIdx.x;
  if (i >= n4) return;
  float4 x = reinterpret_cast<float4*>(a)[i];
  float4 y = reinterpret_cast<const float4*>(b)[i];
  x.x += y.x; x.y += y.y; x.z += y.z; x.w += y.w;
  reinterpret_cast<float4*>(a)[i] = x;
}

__global__ void skip_accum_kernel(float* __restrict__ skip, const float* __restrict__ sa) {
  int i = blockIdx.x * 256 + threadIdx.x;
  if (i >= BATCH * TLEN * DMODEL) return;
  const int d = i % DMODEL;
  const int t = (i / DMODEL) % TLEN;
  const int b = i / (DMODEL * TLEN);
  float s = 0.0f;
  #pragma unroll
  for (int ii = 0; ii < INSTR; ++ii)
    s += sa[(((size_t)(b * INSTR + ii)) * TLEN + t) * DMODEL + d];
  skip[i] += 0.2f * s;
}

__global__ void relumean_kernel(const float* __restrict__ h, float* __restrict__ xm) {
  int i = blockIdx.x * 256 + threadIdx.x;
  if (i >= BATCH * TLEN * DMODEL) return;
  const int d = i % DMODEL;
  const int t = (i / DMODEL) % TLEN;
  const int b = i / (DMODEL * TLEN);
  float s = 0.0f;
  #pragma unroll
  for (int ii = 0; ii < INSTR; ++ii)
    s += fmaxf(h[(((size_t)(b * INSTR + ii)) * TLEN + t) * DMODEL + d], 0.0f);
  xm[i] = 0.2f * s;
}

__global__ void h2g_kernel(const float* __restrict__ h, float* __restrict__ g) {
  const int D4 = DMODEL / 4;
  int i = blockIdx.x * 256 + threadIdx.x;
  if (i >= BI * TLEN * D4) return;
  const int d4  = i % D4;
  const int ins = (i / D4) % INSTR;
  const int t   = (i / (D4 * INSTR)) % TLEN;
  const int b   = i / (D4 * INSTR * TLEN);
  const float4 v = reinterpret_cast<const float4*>(h)
      [(((size_t)(b * INSTR + ins)) * TLEN + t) * D4 + d4];
  reinterpret_cast<float4*>(g)
      [(((size_t)(b * TLEN + t)) * INSTR + ins) * D4 + d4] = v;
}

__global__ void g2h_kernel(const float* __restrict__ g, float* __restrict__ h) {
  const int D4 = DMODEL / 4;
  int i = blockIdx.x * 256 + threadIdx.x;
  if (i >= BI * TLEN * D4) return;
  const int d4  = i % D4;
  const int ins = (i / D4) % INSTR;
  const int t   = (i / (D4 * INSTR)) % TLEN;
  const int b   = i / (D4 * INSTR * TLEN);
  const float4 v = reinterpret_cast<const float4*>(g)
      [(((size_t)(b * TLEN + t)) * INSTR + ins) * D4 + d4];
  reinterpret_cast<float4*>(h)
      [(((size_t)(b * INSTR + ins)) * TLEN + t) * D4 + d4] = v;
}

__global__ void head_main_kernel(const float* __restrict__ xm, const float* __restrict__ w,
                                 const float* __restrict__ b, float* __restrict__ out) {
  int i = blockIdx.x * 256 + threadIdx.x;
  if (i >= BATCH * TLEN * 2) return;
  const int o   = i % 2;
  const int row = i / 2;
  float acc = b[o];
  const float* xr = xm + (size_t)row * DMODEL;
  const float* wr = w + o * DMODEL;
  for (int d = 0; d < DMODEL; ++d) acc += xr[d] * wr[d];
  out[i] = acc;
}

__global__ void skip_treduce_kernel(const float* __restrict__ skip, float* __restrict__ ts) {
  int i = threadIdx.x;                  // single block of 256 covers BATCH*D
  if (i >= BATCH * DMODEL) return;
  const int b = i / DMODEL;
  const int d = i % DMODEL;
  float acc = 0.0f;
  for (int t = 0; t < TLEN; ++t)
    acc += fmaxf(skip[((size_t)b * TLEN + t) * DMODEL + d], 0.0f);
  ts[i] = acc * (1.0f / TLEN);
}

__global__ void head_t_kernel(const float* __restrict__ ts, const float* __restrict__ w,
                              const float* __restrict__ b, float* __restrict__ out) {
  int i = blockIdx.x * 256 + threadIdx.x;
  if (i >= BATCH * 300) return;
  const int o  = i % 300;
  const int bb = i / 300;
  float acc = b[o];
  for (int d = 0; d < DMODEL; ++d) acc += ts[bb * DMODEL + d] * w[o * DMODEL + d];
  out[(size_t)BATCH * TLEN * 2 + i] = acc;
}

// ---------------------------------------------------------------------------
static inline void gemm(const float* A, const float* W, const float* bias,
                        const float* res, float* C, int M, int N, int K, int act,
                        hipStream_t stream) {
  dim3 grid(N / 64, M / 128);
  if (K == 128)
    gemm_wmma_kernel<128><<<grid, dim3(256), 0, stream>>>(A, W, bias, res, C, M, N, act);
  else
    gemm_wmma_kernel<512><<<grid, dim3(256), 0, stream>>>(A, W, bias, res, C, M, N, act);
}

extern "C" void kernel_launch(void* const* d_in, const int* in_sizes, int n_in,
                              void* d_out, int out_size, void* d_ws, size_t ws_size,
                              hipStream_t stream)
{
  (void)in_sizes; (void)n_in; (void)out_size; (void)ws_size;
  const float* x    = (const float*)d_in[0];
  const float* c1w  = (const float*)d_in[1];  const float* c1b = (const float*)d_in[2];
  const float* c2w  = (const float*)d_in[3];  const float* c2b = (const float*)d_in[4];
  const float* c3w  = (const float*)d_in[5];  const float* c3b = (const float*)d_in[6];
  const float* Wq   = (const float*)d_in[7];  const float* bq  = (const float*)d_in[8];
  const float* Wk   = (const float*)d_in[9];  const float* bk  = (const float*)d_in[10];
  const float* Wv   = (const float*)d_in[11]; const float* bvv = (const float*)d_in[12];
  const float* Er   = (const float*)d_in[13];
  const float* ln1s = (const float*)d_in[14]; const float* ln1b = (const float*)d_in[15];
  const float* ln2s = (const float*)d_in[16]; const float* ln2b = (const float*)d_in[17];
  const float* f1w  = (const float*)d_in[18]; const float* f1b  = (const float*)d_in[19];
  const float* f2w  = (const float*)d_in[20]; const float* f2b  = (const float*)d_in[21];
  const float* iinw = (const float*)d_in[22]; const float* iinb = (const float*)d_in[23];
  const float* iow  = (const float*)d_in[24]; const float* iob  = (const float*)d_in[25];
  const float* il1s = (const float*)d_in[26]; const float* il1b = (const float*)d_in[27];
  const float* il2s = (const float*)d_in[28]; const float* il2b = (const float*)d_in[29];
  const float* if1w = (const float*)d_in[30]; const float* if1b = (const float*)d_in[31];
  const float* if2w = (const float*)d_in[32]; const float* if2b = (const float*)d_in[33];
  const float* outw = (const float*)d_in[34]; const float* outb = (const float*)d_in[35];
  const float* otw  = (const float*)d_in[36]; const float* otb  = (const float*)d_in[37];
  float* out = (float*)d_out;

  // Workspace arena (floats). Conv scratch overlays the steady-state buffers:
  //   Union A (27,525,120): conv1 out  |  later q,k,v,sa,g,o (6 x 2,621,440)
  //   Union B (13,107,200): conv2 out  |  later mid (10,485,760) + skip (524,288)
  //   + h (2,621,440) + ln (2,621,440) + ts (256)  => ~184 MB total.
  float* ws = (float*)d_ws;
  const size_t SZ_TOK = (size_t)BI * TLEN * DMODEL;     // 2,621,440
  const size_t UA_SZ  = (size_t)BI * 32 * TLEN * 42;    // 27,525,120
  const size_t MID_SZ = (size_t)MTOK * 512;             // 10,485,760
  const size_t UB_SZ  = (size_t)BI * 64 * TLEN * 10;    // 13,107,200
  float* qb    = ws + 0 * SZ_TOK;
  float* kbuf  = ws + 1 * SZ_TOK;
  float* vbuf  = ws + 2 * SZ_TOK;
  float* sab   = ws + 3 * SZ_TOK;
  float* gb    = ws + 4 * SZ_TOK;
  float* obuf  = ws + 5 * SZ_TOK;
  float* c1buf = ws;                       // overlays qb..obuf (conv stage only)
  float* midb  = ws + UA_SZ;
  float* skipb = midb + MID_SZ;
  float* c2buf = ws + UA_SZ;               // overlays midb+skipb (conv stage only)
  float* hb    = ws + UA_SZ + UB_SZ;
  float* lnb   = hb + SZ_TOK;
  float* tsb   = lnb + SZ_TOK;

  // ---- conv frontend ----
  conv1_kernel<<<(BI*32*TLEN*42)/256, 256, 0, stream>>>(x, c1w, c1b, c1buf);
  conv2_kernel<<<(BI*64*TLEN*10)/256, 256, 0, stream>>>(c1buf, c2w, c2b, c2buf);
  conv3_kernel<<<(BI*TLEN*DMODEL)/256, 256, 0, stream>>>(c2buf, c3w, c3b, hb);
  zero_kernel<<<(BATCH*TLEN*DMODEL)/256, 256, 0, stream>>>(skipb, BATCH*TLEN*DMODEL);

  const int nAttnBlk = (BI * NHEAD * TLEN) / 256;      // 640
  const int nEltBlk  = (int)(SZ_TOK / 4) / 256;        // 2560 (float4 elementwise)
  const int nBT      = (BATCH * TLEN * DMODEL) / 256;  // 2048

  // ---- 9 dilated transformer layers ----
  for (int layer = 0; layer < 9; ++layer) {
    layernorm_kernel<<<MTOK, 128, 0, stream>>>(hb, ln1s + layer*128, ln1b + layer*128, lnb);
    gemm(lnb, Wq + layer*16384, bq + layer*128, nullptr, qb,   MTOK, 128, 128, 0, stream);
    gemm(lnb, Wk + layer*16384, bk + layer*128, nullptr, kbuf, MTOK, 128, 128, 0, stream);
    gemm(lnb, Wv + layer*16384, bvv + layer*128, nullptr, vbuf, MTOK, 128, 128, 0, stream);
    dilated_attn_kernel<<<nAttnBlk, 256, 0, stream>>>(qb, kbuf, vbuf, Er + layer*640, sab, layer);
    add_inplace_kernel<<<nEltBlk, 256, 0, stream>>>(hb, sab, (int)(SZ_TOK/4));
    skip_accum_kernel<<<nBT, 256, 0, stream>>>(skipb, sab);
    layernorm_kernel<<<MTOK, 128, 0, stream>>>(hb, ln2s + layer*128, ln2b + layer*128, lnb);
    gemm(lnb,  f1w + layer*65536, f1b + layer*512, nullptr, midb, MTOK, 512, 128, 2, stream); // gelu
    gemm(midb, f2w + layer*65536, f2b + layer*128, hb,      hb,   MTOK, 128, 512, 0, stream); // +res

    if (layer >= 3 && layer <= 5) {
      const int j = layer - 3;
      h2g_kernel<<<nEltBlk, 256, 0, stream>>>(hb, gb);
      layernorm_kernel<<<MTOK, 128, 0, stream>>>(gb, il1s + j*128, il1b + j*128, lnb);
      gemm(lnb, iinw + j*49152, iinb + j*384, nullptr, midb, MTOK, 384, 128, 0, stream);
      instr_attn_kernel<<<nAttnBlk, 256, 0, stream>>>(midb, obuf);
      gemm(obuf, iow + j*16384, iob + j*128, gb, gb, MTOK, 128, 128, 0, stream);
      layernorm_kernel<<<MTOK, 128, 0, stream>>>(gb, il2s + j*128, il2b + j*128, lnb);
      gemm(lnb,  if1w + j*65536, if1b + j*512, nullptr, midb, MTOK, 512, 128, 1, stream); // relu
      gemm(midb, if2w + j*65536, if2b + j*128, gb,      gb,   MTOK, 128, 512, 0, stream); // +res
      g2h_kernel<<<nEltBlk, 256, 0, stream>>>(gb, hb);
    }
  }

  // ---- heads ----
  relumean_kernel<<<nBT, 256, 0, stream>>>(hb, lnb);             // xm = mean_i relu(h)
  head_main_kernel<<<(BATCH*TLEN*2)/256, 256, 0, stream>>>(lnb, outw, outb, out);
  skip_treduce_kernel<<<1, 256, 0, stream>>>(skipb, tsb);        // mean_t relu(sum skips)
  head_t_kernel<<<(BATCH*300 + 255)/256, 256, 0, stream>>>(tsb, otw, otb, out);
}
